// MLGA_63711544868892
// MI455X (gfx1250) — matrix operand order repaired and checked by reading, not compiled, not run on hardware
//
#include <hip/hip_runtime.h>
#include <cstdint>
#include <cstddef>

typedef _Float16 half_t;
typedef __attribute__((ext_vector_type(16))) _Float16 v16h;
typedef __attribute__((ext_vector_type(8)))  float    v8f;

#define NB   8
#define NG   2048
#define NK   24
#define NC   96
#define NGK  (NG*NK)
#define BG   (NB*NG)
#define EPSB 1e-5f

// LDS tile strides in halves (33 dwords -> conflict-free on 64 banks)
#define ASTR 66
#define BSTR 66

// ---------------------------------------------------------------------------
// Generic WMMA GEMM:  Y[b,row,col] = sum_k A[b,row,k] * B(b,k,col)  (+bias)
// A: f32 strided view (weights or activations).  B: Loader (applies previous
// stage's BatchNorm + activation on load).  Epilogue optionally accumulates
// per-row sum/sumsq for the NEXT BatchNorm via LDS ds_add_f32 + global
// float atomics.  Block = 128 threads = 4 wave32; K-chunk = 64 -> 2 WMMAs
// per chunk; B tile is column-major in LDS so fragment gathers are b32 reads.
// ---------------------------------------------------------------------------
struct AView { const float* p; long sb, sr, sk; };

struct LdHalf {              // f16 (Kc,N) per batch, optional BN + activation
  const half_t* X; long sb; int ncols;
  const float* ms; const float* g; const float* be; int act;
  __device__ __forceinline__ half_t load(int b, int k, int c) const {
    float v = (float)X[(long)b*sb + (long)k*ncols + c];
    if (ms) v = (v - ms[2*k]) * ms[2*k+1];
    if (g)  v = v * g[k] + be[k];
    if (act == 1)      v = v > 0.f ? v : 0.f;
    else if (act == 2) v = v > 0.f ? v : 0.2f*v;
    return (half_t)v;
  }
  __device__ __forceinline__ void pref(int b, int k, int c) const {
    __builtin_prefetch(&X[(long)b*sb + (long)k*ncols + c], 0, 1);
  }
};

struct LdF32 {               // plain f32 strided view
  const float* X; long sb, sk, sc;
  __device__ __forceinline__ half_t load(int b, int k, int c) const {
    return (half_t)X[(long)b*sb + (long)k*sk + (long)c*sc];
  }
  __device__ __forceinline__ void pref(int, int, int) const {}
};

struct LdKX {                // gather kx = concat(knn_x, broadcast lc_x), (192, G*K)
  const float* knn_x; const float* lc_x;
  __device__ __forceinline__ half_t load(int b, int k, int c) const {
    const int g = c / NK, kk = c - g*NK;
    float v;
    if (k < NC) v = knn_x[(((long)b*NG + g)*NK + kk)*NC + k];
    else        v = lc_x [((long)b*NG + g)*NC + (k - NC)];
    return (half_t)v;
  }
  __device__ __forceinline__ void pref(int, int, int) const {}
};

template <class L>
__global__ void __launch_bounds__(128)
gemm_k(AView A, L ld, int M, int Kc, int N,
       const float* bias, half_t* y16, float* y32, float* stat, long ysb)
{
  __shared__ half_t At[16*ASTR];
  __shared__ half_t Bt[64*BSTR];
  __shared__ float  sS[16], sQ[16];
  const int tid = threadIdx.x;
  const int b   = blockIdx.z;
  const int m0  = blockIdx.y * 16;
  const int c0  = blockIdx.x * 64;
  if (stat && tid < 16) { sS[tid] = 0.f; sQ[tid] = 0.f; }

  v8f acc = {0.f,0.f,0.f,0.f,0.f,0.f,0.f,0.f};
  for (int k0 = 0; k0 < Kc; k0 += 64) {
    __syncthreads();
    const bool full = (k0 + 64 <= Kc) & (m0 + 16 <= M) & (c0 + 64 <= N);
    // ---- stage A tile: 16 rows x 64 k (row-major, stride ASTR) ----
    {
      const int m = tid >> 3, kk = (tid & 7) * 8;
      const int r = m0 + m;
      if (full) {
        const float* ap = A.p + (long)b*A.sb + (long)r*A.sr + (long)(k0 + kk)*A.sk;
#pragma unroll
        for (int u = 0; u < 8; ++u)
          At[m*ASTR + kk + u] = (half_t)ap[(long)u*A.sk];
      } else {
#pragma unroll
        for (int u = 0; u < 8; ++u) {
          const int k = k0 + kk + u;
          At[m*ASTR + kk + u] = (r < M && k < Kc)
              ? (half_t)A.p[(long)b*A.sb + (long)r*A.sr + (long)k*A.sk]
              : (half_t)0.f;
        }
      }
    }
    // ---- stage B tile: 64 k x 64 cols, COLUMN-major in LDS (stride BSTR) --
    {
      const int c  = tid & 63;
      const int kb = (tid >> 6) * 32;
      if (full) {
        if (k0 + 64 < Kc) ld.pref(b, k0 + 64 + kb, c0 + c);   // next chunk
#pragma unroll 8
        for (int u = 0; u < 32; ++u)
          Bt[c*BSTR + kb + u] = ld.load(b, k0 + kb + u, c0 + c);
      } else {
        const int cc = c0 + c;
#pragma unroll 8
        for (int u = 0; u < 32; ++u) {
          const int k = k0 + kb + u;
          Bt[c*BSTR + kb + u] = (k < Kc && cc < N) ? ld.load(b, k, cc)
                                                   : (half_t)0.f;
        }
      }
    }
    __syncthreads();
    // ---- two 16x16x32 WMMAs; fragments gathered as b32 LDS reads ----
    const int l = tid & 31, mn = l & 15, hs = l >> 4;
    const int co = (tid >> 5) * 16;
#pragma unroll
    for (int sub = 0; sub < 2; ++sub) {
      union { v16h v; uint32_t u[8]; } fa, fb;
#pragma unroll
      for (int j = 0; j < 8; ++j) {
        // A layout (ISA 7.12.2, 16-bit A 16x32): pairs at even k offsets
        const int ka = sub*32 + ((j & 4) << 2) + hs*8 + ((j & 3) << 1);
        fa.u[j] = *(const uint32_t*)(At + mn*ASTR + ka);
        // B layout: same k order as A (lanes 0-15 K=0..7,16..23; lanes 16-31 K=8..15,24..31) (col-major)
        const int kb2 = sub*32 + ((j & 4) << 2) + hs*8 + ((j & 3) << 1);
        fb.u[j] = *(const uint32_t*)(Bt + (co + mn)*BSTR + kb2);
      }
      acc = __builtin_amdgcn_wmma_f32_16x16x32_f16(false, fa.v, false, fb.v,
                                                   (short)0, acc, false, false);
    }
  }

  const int l   = tid & 31;
  const int col = c0 + (tid >> 5)*16 + (l & 15);
  const int mb  = (l >> 4) * 8;
#pragma unroll
  for (int r = 0; r < 8; ++r) {
    const int row = m0 + mb + r;
    if (row < M && col < N) {
      float v = acc[r];
      if (bias) v += bias[row];
      const long o = (long)b*ysb + (long)row*N + col;
      if (y16) y16[o] = (half_t)v;
      if (y32) y32[o] = v;
      if (stat) { atomicAdd(&sS[mb + r], v); atomicAdd(&sQ[mb + r], v*v); }
    }
  }
  if (stat) {
    __syncthreads();
    if (tid < 16 && (m0 + tid) < M) {
      atomicAdd(&stat[2*(m0 + tid)],     sS[tid]);
      atomicAdd(&stat[2*(m0 + tid) + 1], sQ[tid]);
    }
  }
}

// ------------------------------- small kernels -----------------------------
__global__ void zero_kernel(float* p, int n) {
  int i = blockIdx.x*256 + threadIdx.x;
  if (i < n) p[i] = 0.f;
}

__global__ void finstats_kernel(const float* acc, float* fin, int Cn, float cnt) {
  int c = blockIdx.x*256 + threadIdx.x;
  if (c >= Cn) return;
  float m = acc[2*c] / cnt;
  float v = acc[2*c+1] / cnt - m*m;
  fin[2*c]   = m;
  fin[2*c+1] = rsqrtf(fmaxf(v, 0.f) + EPSB);
}

__global__ void stdacc_kernel(const float* knn_xyz, const float* lc_xyz, float* acc) {
  __shared__ float s1[256], s2[256];
  const long n = (long)BG * NK * 3;
  float a = 0.f, q = 0.f;
  for (long i = (long)blockIdx.x*256 + threadIdx.x; i < n; i += (long)gridDim.x*256) {
    const long bg = i / (NK*3);
    const int  d3 = (int)(i % 3);
    const float d = knn_xyz[i] - lc_xyz[bg*3 + d3];
    a += d; q += d*d;
  }
  s1[threadIdx.x] = a; s2[threadIdx.x] = q; __syncthreads();
  for (int s = 128; s > 0; s >>= 1) {
    if (threadIdx.x < s) { s1[threadIdx.x] += s1[threadIdx.x+s]; s2[threadIdx.x] += s2[threadIdx.x+s]; }
    __syncthreads();
  }
  if (threadIdx.x == 0) { atomicAdd(&acc[0], s1[0]); atomicAdd(&acc[1], s2[0]); }
}

__global__ void stdfin_kernel(const float* acc, float* inv_std) {
  const float n = (float)((long)BG * NK * 3);
  const float mean = acc[0] / n;
  const float var  = (acc[1] - n*mean*mean) / (n - 1.f);
  inv_std[0] = 1.f / (sqrtf(fmaxf(var, 0.f)) + 1e-5f);
}

// closed-form symmetric 3x3 eigen-decomposition of (c c^T - E[pp^T])
__global__ void geo_kernel(const float* knn_xyz, float* nrm, float* curv) {
  long idx = (long)blockIdx.x*256 + threadIdx.x;
  if (idx >= (long)BG) return;
  const float* P = knn_xyz + idx*NK*3;
  float cx=0,cy=0,cz=0, xx=0,xy=0,xz=0,yy=0,yz=0,zz=0;
  for (int i = 0; i < 16; ++i) {
    float x=P[i*3], y=P[i*3+1], z=P[i*3+2];
    cx+=x; cy+=y; cz+=z;
    xx+=x*x; xy+=x*y; xz+=x*z; yy+=y*y; yz+=y*z; zz+=z*z;
  }
  const float is = 1.f/16.f;
  cx*=is; cy*=is; cz*=is;
  float a00=cx*cx-xx*is, a11=cy*cy-yy*is, a22=cz*cz-zz*is;
  float a01=cx*cy-xy*is, a02=cx*cz-xz*is, a12=cy*cz-yz*is;
  const float qd = (a00+a11+a22)/3.f;
  const float p1 = a01*a01 + a02*a02 + a12*a12;
  const float p2 = (a00-qd)*(a00-qd)+(a11-qd)*(a11-qd)+(a22-qd)*(a22-qd)+2.f*p1;
  const float pp = sqrtf(p2/6.f + 1e-30f);
  const float ip = 1.f/pp;
  const float b00=(a00-qd)*ip, b11=(a11-qd)*ip, b22=(a22-qd)*ip;
  const float b01=a01*ip, b02=a02*ip, b12=a12*ip;
  float detB = b00*(b11*b22-b12*b12) - b01*(b01*b22-b12*b02) + b02*(b01*b12-b11*b02);
  float r = fminf(1.f, fmaxf(-1.f, 0.5f*detB));
  const float phi = acosf(r)/3.f;
  const float e1 = qd + 2.f*pp*cosf(phi);
  const float e3 = qd + 2.f*pp*cosf(phi + 2.09439510239319549f);
  const float e2 = 3.f*qd - e1 - e3;
  float s0=fabsf(e3), s1=fabsf(e2), s2=fabsf(e1);      // desc for NSD matrix
  if (s0 < s1) { float t=s0; s0=s1; s1=t; }
  if (s1 < s2) { float t=s1; s1=s2; s2=t; }
  if (s0 < s1) { float t=s0; s0=s1; s1=t; }
  const float cs = 1.f/(s0+s1+s2+3e-9f);
  curv[idx*3+0]=(s0+1e-9f)*cs; curv[idx*3+1]=(s1+1e-9f)*cs; curv[idx*3+2]=(s2+1e-9f)*cs;
  // eigenvector of smallest singular value -> eigenvalue e1
  const float r0x=a00-e1, r0y=a01, r0z=a02;
  const float r1x=a01, r1y=a11-e1, r1z=a12;
  const float r2x=a02, r2y=a12, r2z=a22-e1;
  float v0x=r0y*r1z-r0z*r1y, v0y=r0z*r1x-r0x*r1z, v0z=r0x*r1y-r0y*r1x;
  float v1x=r0y*r2z-r0z*r2y, v1y=r0z*r2x-r0x*r2z, v1z=r0x*r2y-r0y*r2x;
  float v2x=r1y*r2z-r1z*r2y, v2y=r1z*r2x-r1x*r2z, v2z=r1x*r2y-r1y*r2x;
  float n0=v0x*v0x+v0y*v0y+v0z*v0z, n1=v1x*v1x+v1y*v1y+v1z*v1z, n2=v2x*v2x+v2y*v2y+v2z*v2z;
  float vx=v0x, vy=v0y, vz=v0z, nn=n0;
  if (n1 > nn) { vx=v1x; vy=v1y; vz=v1z; nn=n1; }
  if (n2 > nn) { vx=v2x; vy=v2y; vz=v2z; nn=n2; }
  if (nn < 1e-20f) { vx=0.f; vy=0.f; vz=1.f; nn=1.f; }
  const float rn = rsqrtf(nn);
  nrm[idx*3+0]=vx*rn; nrm[idx*3+1]=vy*rn; nrm[idx*3+2]=vz*rn;
}

__global__ void pool_kernel(const half_t* h3, const float* fin, const float* gg,
                            const float* bb, float* pooled) {
  const int c = blockIdx.x, b = blockIdx.y;
  __shared__ float red[256];
  const float m = fin[2*c], rs = fin[2*c+1], ga = gg[c], be = bb[c];
  float best = -3.0e38f;
  for (int g = threadIdx.x; g < NG; g += 256) {
    float v = (float)h3[((long)b*1024 + c)*NG + g];
    v = (v - m)*rs*ga + be;
    v = v >= 0.f ? v : 0.2f*v;
    best = fmaxf(best, v);
  }
  red[threadIdx.x] = best; __syncthreads();
  for (int s = 128; s > 0; s >>= 1) {
    if (threadIdx.x < s) red[threadIdx.x] = fmaxf(red[threadIdx.x], red[threadIdx.x+s]);
    __syncthreads();
  }
  if (threadIdx.x == 0) pooled[b*1024 + c] = red[0];
}

// t-net FC chain (B=8 only): l1(512) -> l2(256) -> tr(9), BN over batch axis.
__global__ void __launch_bounds__(512) fc_kernel(
    const float* pooled, const float* l1w, const float* l1g, const float* l1b,
    const float* l2w, const float* l2g, const float* l2b,
    const float* trw, const float* trb, float* tmat)
{
  __shared__ float h1[8][512];
  __shared__ float h2[8][256];
  const int t = threadIdx.x;
  {
    for (int b = 0; b < 8; ++b) {
      float a = 0.f;
      for (int k = 0; k < 1024; ++k) a += l1w[t*1024 + k] * pooled[b*1024 + k];
      h1[b][t] = a;
    }
    float m = 0.f; for (int b = 0; b < 8; ++b) m += h1[b][t]; m *= 0.125f;
    float v = 0.f; for (int b = 0; b < 8; ++b) { float d = h1[b][t]-m; v += d*d; } v *= 0.125f;
    const float rs = rsqrtf(v + EPSB);
    for (int b = 0; b < 8; ++b) {
      float x = (h1[b][t]-m)*rs*l1g[t] + l1b[t];
      h1[b][t] = x >= 0.f ? x : 0.2f*x;
    }
  }
  __syncthreads();
  if (t < 256) {
    for (int b = 0; b < 8; ++b) {
      float a = 0.f;
      for (int k = 0; k < 512; ++k) a += l2w[t*512 + k] * h1[b][k];
      h2[b][t] = a;
    }
    float m = 0.f; for (int b = 0; b < 8; ++b) m += h2[b][t]; m *= 0.125f;
    float v = 0.f; for (int b = 0; b < 8; ++b) { float d = h2[b][t]-m; v += d*d; } v *= 0.125f;
    const float rs = rsqrtf(v + EPSB);
    for (int b = 0; b < 8; ++b) {
      float x = (h2[b][t]-m)*rs*l2g[t] + l2b[t];
      h2[b][t] = x >= 0.f ? x : 0.2f*x;
    }
  }
  __syncthreads();
  if (t < 72) {
    const int b = t / 9, j = t % 9;
    float a = trb[j];
    for (int k = 0; k < 256; ++k) a += trw[j*256 + k] * h2[b][k];
    tmat[t] = a;
  }
}

__global__ void applyt_kernel(const float* nrm, const float* tmat, float* n_t) {
  long idx = (long)blockIdx.x*256 + threadIdx.x;
  if (idx >= (long)BG) return;
  const int b = (int)(idx / NG);
  const float x = nrm[idx*3], y = nrm[idx*3+1], z = nrm[idx*3+2];
  const float* tm = tmat + b*9;
  n_t[idx*3+0] = x*tm[0] + y*tm[3] + z*tm[6] + x;
  n_t[idx*3+1] = x*tm[1] + y*tm[4] + z*tm[7] + y;
  n_t[idx*3+2] = x*tm[2] + y*tm[5] + z*tm[8] + z;
}

// w0 = (relu(bn(y1)) + pe) * pe, pe computed on the fly (sin/cos -> TRANS32)
__global__ void pew_kernel(const half_t* y1, const float* fin, const float* g,
                           const float* be, const float* knn_xyz, const float* lc_xyz,
                           const float* inv_std, half_t* w0) {
  const long total = (long)NB*NC*NG*NK;
  for (long idx = (long)blockIdx.x*256 + threadIdx.x; idx < total;
       idx += (long)gridDim.x*256) {
    const int k = (int)(idx % NK);
    long t1 = idx / NK; const int gg = (int)(t1 % NG);
    long t2 = t1 / NG; const int c = (int)(t2 % NC); const int b = (int)(t2 / NC);
    const int d3 = c >> 5, tt = c & 31, j = tt & 15;
    const long bg = (long)b*NG + gg;
    const float xyz = (knn_xyz[(bg*NK + k)*3 + d3] - lc_xyz[bg*3 + d3]) * inv_std[0];
    const float freq = 100.f * __powf(1000.f, -(float)j * 0.0625f);
    const float dv = xyz * freq;
    const float pe = (tt < 16) ? __sinf(dv) : __cosf(dv);
    float y = (float)y1[idx];
    y = (y - fin[2*c]) * fin[2*c+1] * g[c] + be[c];
    y = y > 0.f ? y : 0.f;
    w0[idx] = (half_t)((y + pe) * pe);
  }
}

// wout = relu(bn(h2raw) + wres)
__global__ void resid_kernel(const half_t* h2raw, const float* fin, const float* g,
                             const float* be, const half_t* wres, half_t* wout) {
  const long total = (long)NB*NC*NGK;
  for (long idx = (long)blockIdx.x*256 + threadIdx.x; idx < total;
       idx += (long)gridDim.x*256) {
    const int c = (int)((idx / NGK) % NC);
    float v = ((float)h2raw[idx] - fin[2*c]) * fin[2*c+1] * g[c] + be[c];
    v += (float)wres[idx];
    wout[idx] = (half_t)(v > 0.f ? v : 0.f);
  }
}

// s = a0*(max_k - mean_k)(w) + b0 + a1*relu(bn(est_n)) + b1 + a2*relu(bn(est_c)) + b2
__global__ void combines_kernel(const half_t* w,
    const half_t* estn, const float* finn, const float* gn, const float* bn,
    const half_t* estc, const float* finc, const float* gc, const float* bc,
    const float* maa_a, const float* maa_b, float* s32, half_t* s16) {
  const long total = (long)NB*NC*NG;
  for (long idx = (long)blockIdx.x*256 + threadIdx.x; idx < total;
       idx += (long)gridDim.x*256) {
    const int c = (int)((idx / NG) % NC);
    const half_t* p = w + idx*NK;
    float mx = -3.0e38f, mean = 0.f;
    for (int k = 0; k < NK; ++k) { float v = (float)p[k]; mx = fmaxf(mx, v); mean += v; }
    mean *= (1.f/NK);
    const float lc = mx - mean;
    float en = ((float)estn[idx] - finn[2*c]) * finn[2*c+1] * gn[c] + bn[c];
    en = en > 0.f ? en : 0.f;
    float ec = ((float)estc[idx] - finc[2*c]) * finc[2*c+1] * gc[c] + bc[c];
    ec = ec > 0.f ? ec : 0.f;
    const float sv = maa_a[c]*lc + maa_b[c]
                   + maa_a[NC + c]*en + maa_b[NC + c]
                   + maa_a[2*NC + c]*ec + maa_b[2*NC + c];
    s32[idx] = sv; s16[idx] = (half_t)sv;
  }
}

__global__ void softmax_kernel(half_t* attn) {
  const int n = blockIdx.x, b = blockIdx.y;
  half_t* row = attn + ((long)b*NG + n)*NG;
  __shared__ float red[256];
  const int t = threadIdx.x;
  float mx = -3.0e38f;
  for (int m = t; m < NG; m += 256) mx = fmaxf(mx, (float)row[m]);
  red[t] = mx; __syncthreads();
  for (int s = 128; s > 0; s >>= 1) { if (t < s) red[t] = fmaxf(red[t], red[t+s]); __syncthreads(); }
  mx = red[0]; __syncthreads();
  float sum = 0.f;
  for (int m = t; m < NG; m += 256) sum += __expf((float)row[m] - mx);
  red[t] = sum; __syncthreads();
  for (int s = 128; s > 0; s >>= 1) { if (t < s) red[t] += red[t+s]; __syncthreads(); }
  const float inv = 1.f / red[0]; __syncthreads();
  for (int m = t; m < NG; m += 256)
    row[m] = (half_t)(__expf((float)row[m] - mx) * inv);
}

__global__ void colsum_kernel(const half_t* attn, float* colsum) {
  const int m = blockIdx.x*256 + threadIdx.x;
  const int b = blockIdx.y;
  if (m >= NG) return;
  const half_t* base = attn + (long)b*NG*NG + m;
  float a = 0.f;
  for (int n = 0; n < NG; ++n) a += (float)base[(long)n*NG];
  colsum[b*NG + m] = a;
}

__global__ void scaleattn_kernel(half_t* attn, const float* colsum) {
  const long total = (long)NB*NG*NG;
  for (long idx = (long)blockIdx.x*256 + threadIdx.x; idx < total;
       idx += (long)gridDim.x*256) {
    const int m = (int)(idx % NG);
    const int b = (int)(idx / ((long)NG*NG));
    attn[idx] = (half_t)((float)attn[idx] / (1e-9f + colsum[b*NG + m]));
  }
}

__global__ void diff_kernel(const float* s32, const float* xr32, half_t* d16) {
  const long total = (long)NB*NC*NG;
  for (long idx = (long)blockIdx.x*256 + threadIdx.x; idx < total;
       idx += (long)gridDim.x*256)
    d16[idx] = (half_t)(s32[idx] - xr32[idx]);
}

__global__ void final_kernel(const float* s32, const half_t* trraw, const float* fin,
                             const float* g, const float* be, float* out) {
  const long total = (long)NB*NC*NG;
  for (long idx = (long)blockIdx.x*256 + threadIdx.x; idx < total;
       idx += (long)gridDim.x*256) {
    const int c = (int)((idx / NG) % NC);
    float v = ((float)trraw[idx] - fin[2*c]) * fin[2*c+1] * g[c] + be[c];
    v = v > 0.f ? v : 0.f;
    out[idx] = s32[idx] + v;
  }
}

// ---------------------------------------------------------------------------
// Host side
// ---------------------------------------------------------------------------
enum {
  IN_LC_XYZ = 0, IN_LC_X, IN_KNN_XYZ, IN_KNN_X,
  IN_LIN1_W, IN_LIN1_G, IN_LIN1_B,
  IN_NORM_W, IN_NORM_G, IN_NORM_B,
  IN_CURV_W, IN_CURV_G, IN_CURV_B,
  IN_B0_W1, IN_B0_B1, IN_B0_G1, IN_B0_BE1, IN_B0_W2, IN_B0_B2, IN_B0_G2, IN_B0_BE2,
  IN_B1_W1, IN_B1_B1, IN_B1_G1, IN_B1_BE1, IN_B1_W2, IN_B1_B2, IN_B1_G2, IN_B1_BE2,
  IN_MAA_A, IN_MAA_B,
  IN_C1_W, IN_C1_G, IN_C1_B, IN_C2_W, IN_C2_G, IN_C2_B, IN_C3_W, IN_C3_G, IN_C3_B,
  IN_L1_W, IN_L1_G, IN_L1_B, IN_L2_W, IN_L2_G, IN_L2_B, IN_TRN_W, IN_TRN_B,
  IN_QK_W, IN_V_W, IN_V_B, IN_SA_TR_W, IN_SA_TR_B, IN_SA_G, IN_SA_BE
};

// stat accumulator float offsets (pairs: sum, sumsq per channel)
#define SO_C1  0
#define SO_C2  128
#define SO_C3  384
#define SO_EN  2432
#define SO_EC  2624
#define SO_L1  2816
#define SO_B1A 3008
#define SO_B1B 3104
#define SO_B2A 3296
#define SO_B2B 3392
#define SO_SA  3584
#define SO_STD 3776
#define STATN  4096

static inline dim3 gemm_grid(int M, int N) { return dim3((N + 63)/64, (M + 15)/16, NB); }
static inline unsigned gs_blocks(long total) {
  long b = (total + 255) / 256;
  if (b > 16384) b = 16384;
  return (unsigned)b;
}

extern "C" void kernel_launch(void* const* d_in, const int* in_sizes, int n_in,
                              void* d_out, int out_size, void* d_ws, size_t ws_size,
                              hipStream_t stream) {
  (void)in_sizes; (void)n_in; (void)out_size; (void)ws_size;
  const float* lc_xyz  = (const float*)d_in[IN_LC_XYZ];
  const float* lc_x    = (const float*)d_in[IN_LC_X];
  const float* knn_xyz = (const float*)d_in[IN_KNN_XYZ];
  const float* knn_x   = (const float*)d_in[IN_KNN_X];
  auto F = [&](int i) { return (const float*)d_in[i]; };
  float* out = (float*)d_out;

  // ---- workspace carve-up ----
  char* W = (char*)d_ws; size_t off = 0;
  auto ALLOC = [&](size_t bytes) -> void* {
    void* p = W + off; off += (bytes + 255) & ~(size_t)255; return p;
  };
  float*  acc   = (float*) ALLOC((size_t)STATN*4);
  float*  fin   = (float*) ALLOC((size_t)STATN*4);
  float*  nrm   = (float*) ALLOC((size_t)BG*3*4);
  float*  curv  = (float*) ALLOC((size_t)BG*3*4);
  float*  n_t   = (float*) ALLOC((size_t)BG*3*4);
  half_t* h1raw = (half_t*)ALLOC((size_t)NB*64*NG*2);
  half_t* h2raw = (half_t*)ALLOC((size_t)NB*128*NG*2);
  half_t* h3raw = (half_t*)ALLOC((size_t)NB*1024*NG*2);
  float*  pooled= (float*) ALLOC((size_t)NB*1024*4);
  float*  tmat  = (float*) ALLOC(72*4);
  half_t* estn  = (half_t*)ALLOC((size_t)NB*NC*NG*2);
  half_t* estc  = (half_t*)ALLOC((size_t)NB*NC*NG*2);
  float*  s32   = (float*) ALLOC((size_t)NB*NC*NG*4);
  half_t* s16   = (half_t*)ALLOC((size_t)NB*NC*NG*2);
  half_t* xk16  = (half_t*)ALLOC((size_t)NB*24*NG*2);
  float*  xk32  = (float*) ALLOC((size_t)NB*24*NG*4);
  float*  xv32  = (float*) ALLOC((size_t)NB*NC*NG*4);
  float*  xr32  = (float*) ALLOC((size_t)NB*NC*NG*4);
  half_t* dsr16 = (half_t*)ALLOC((size_t)NB*NC*NG*2);
  half_t* trraw = (half_t*)ALLOC((size_t)NB*NC*NG*2);
  float*  colsum= (float*) ALLOC((size_t)BG*4);
  half_t* bufP  = (half_t*)ALLOC((size_t)NB*NC*NGK*2);
  half_t* bufQ  = (half_t*)ALLOC((size_t)NB*NC*NGK*2);
  half_t* hA    = (half_t*)ALLOC((size_t)NB*48*NGK*2);
  half_t* hB    = (half_t*)ALLOC((size_t)NB*NC*NGK*2);
  half_t* attn  = (half_t*)ALLOC((size_t)NB*NG*NG*2);

  const float CNT_BG  = (float)BG;
  const float CNT_BGK = (float)((long)NB*NGK);
  auto FINAL = [&](int so, int C, float cnt) {
    finstats_kernel<<<(C+255)/256, 256, 0, stream>>>(acc + so, fin + so, C, cnt);
  };

  // 0) zero stat accumulators; global std of d
  zero_kernel<<<(STATN+255)/256, 256, 0, stream>>>(acc, STATN);
  stdacc_kernel<<<1024, 256, 0, stream>>>(knn_xyz, lc_xyz, acc + SO_STD);
  stdfin_kernel<<<1, 1, 0, stream>>>(acc + SO_STD, fin + SO_STD);

  // 1) local geometry: normals + curvature
  geo_kernel<<<(BG+255)/256, 256, 0, stream>>>(knn_xyz, nrm, curv);

  // 2) t-net conv chain (WMMA GEMMs, BN stats fused in epilogue)
  gemm_k<LdF32><<<gemm_grid(64, NG), 128, 0, stream>>>(
      AView{F(IN_C1_W), 0, 3, 1}, LdF32{nrm, (long)NG*3, 1, 3},
      64, 3, NG, nullptr, h1raw, nullptr, acc + SO_C1, (long)64*NG);
  FINAL(SO_C1, 64, CNT_BG);
  gemm_k<LdHalf><<<gemm_grid(128, NG), 128, 0, stream>>>(
      AView{F(IN_C2_W), 0, 64, 1},
      LdHalf{h1raw, (long)64*NG, NG, fin + SO_C1, F(IN_C1_G), F(IN_C1_B), 2},
      128, 64, NG, nullptr, h2raw, nullptr, acc + SO_C2, (long)128*NG);
  FINAL(SO_C2, 128, CNT_BG);
  gemm_k<LdHalf><<<gemm_grid(1024, NG), 128, 0, stream>>>(
      AView{F(IN_C3_W), 0, 128, 1},
      LdHalf{h2raw, (long)128*NG, NG, fin + SO_C2, F(IN_C2_G), F(IN_C2_B), 2},
      1024, 128, NG, nullptr, h3raw, nullptr, acc + SO_C3, (long)1024*NG);
  FINAL(SO_C3, 1024, CNT_BG);
  pool_kernel<<<dim3(1024, NB), 256, 0, stream>>>(h3raw, fin + SO_C3, F(IN_C3_G), F(IN_C3_B), pooled);
  fc_kernel<<<1, 512, 0, stream>>>(pooled, F(IN_L1_W), F(IN_L1_G), F(IN_L1_B),
                                   F(IN_L2_W), F(IN_L2_G), F(IN_L2_B),
                                   F(IN_TRN_W), F(IN_TRN_B), tmat);
  applyt_kernel<<<(BG+255)/256, 256, 0, stream>>>(nrm, tmat, n_t);

  // 3) est_n / est_c heads (pre-BN stored; relu(bn) applied at combine)
  gemm_k<LdF32><<<gemm_grid(NC, NG), 128, 0, stream>>>(
      AView{F(IN_NORM_W), 0, 3, 1}, LdF32{n_t, (long)NG*3, 1, 3},
      NC, 3, NG, nullptr, estn, nullptr, acc + SO_EN, (long)NC*NG);
  FINAL(SO_EN, NC, CNT_BG);
  gemm_k<LdF32><<<gemm_grid(NC, NG), 128, 0, stream>>>(
      AView{F(IN_CURV_W), 0, 3, 1}, LdF32{curv, (long)NG*3, 1, 3},
      NC, 3, NG, nullptr, estc, nullptr, acc + SO_EC, (long)NC*NG);
  FINAL(SO_EC, NC, CNT_BG);

  // 4) lin1 on gathered kx (96 x 192 x 49152 per batch)
  gemm_k<LdKX><<<gemm_grid(NC, NGK), 128, 0, stream>>>(
      AView{F(IN_LIN1_W), 0, 192, 1}, LdKX{knn_x, lc_x},
      NC, 192, NGK, nullptr, bufP, nullptr, acc + SO_L1, (long)NC*NGK);
  FINAL(SO_L1, NC, CNT_BGK);

  // 5) positional embedding fuse: w0 = (relu(bn(y1)) + pe) * pe
  pew_kernel<<<gs_blocks((long)NB*NC*NGK), 256, 0, stream>>>(
      bufP, fin + SO_L1, F(IN_LIN1_G), F(IN_LIN1_B),
      knn_xyz, lc_xyz, fin + SO_STD, bufQ);

  // 6) residual block 1: bufQ -> bufP
  gemm_k<LdHalf><<<gemm_grid(48, NGK), 128, 0, stream>>>(
      AView{F(IN_B0_W1), 0, NC, 1},
      LdHalf{bufQ, (long)NC*NGK, NGK, nullptr, nullptr, nullptr, 0},
      48, NC, NGK, F(IN_B0_B1), hA, nullptr, acc + SO_B1A, (long)48*NGK);
  FINAL(SO_B1A, 48, CNT_BGK);
  gemm_k<LdHalf><<<gemm_grid(NC, NGK), 128, 0, stream>>>(
      AView{F(IN_B0_W2), 0, 48, 1},
      LdHalf{hA, (long)48*NGK, NGK, fin + SO_B1A, F(IN_B0_G1), F(IN_B0_BE1), 1},
      NC, 48, NGK, F(IN_B0_B2), hB, nullptr, acc + SO_B1B, (long)NC*NGK);
  FINAL(SO_B1B, NC, CNT_BGK);
  resid_kernel<<<gs_blocks((long)NB*NC*NGK), 256, 0, stream>>>(
      hB, fin + SO_B1B, F(IN_B0_G2), F(IN_B0_BE2), bufQ, bufP);

  // 7) residual block 2: bufP -> bufQ
  gemm_k<LdHalf><<<gemm_grid(48, NGK), 128, 0, stream>>>(
      AView{F(IN_B1_W1), 0, NC, 1},
      LdHalf{bufP, (long)NC*NGK, NGK, nullptr, nullptr, nullptr, 0},
      48, NC, NGK, F(IN_B1_B1), hA, nullptr, acc + SO_B2A, (long)48*NGK);
  FINAL(SO_B2A, 48, CNT_BGK);
  gemm_k<LdHalf><<<gemm_grid(NC, NGK), 128, 0, stream>>>(
      AView{F(IN_B1_W2), 0, 48, 1},
      LdHalf{hA, (long)48*NGK, NGK, fin + SO_B2A, F(IN_B1_G1), F(IN_B1_BE1), 1},
      NC, 48, NGK, F(IN_B1_B2), hB, nullptr, acc + SO_B2B, (long)NC*NGK);
  FINAL(SO_B2B, NC, CNT_BGK);
  resid_kernel<<<gs_blocks((long)NB*NC*NGK), 256, 0, stream>>>(
      hB, fin + SO_B2B, F(IN_B1_G2), F(IN_B1_BE2), bufP, bufQ);

  // 8) combine max-mean pooling with est heads + maa
  combines_kernel<<<gs_blocks((long)NB*NC*NG), 256, 0, stream>>>(
      bufQ, estn, fin + SO_EN, F(IN_NORM_G), F(IN_NORM_B),
      estc, fin + SO_EC, F(IN_CURV_G), F(IN_CURV_B),
      F(IN_MAA_A), F(IN_MAA_B), s32, s16);

  // 9) self-attention
  gemm_k<LdHalf><<<gemm_grid(24, NG), 128, 0, stream>>>(   // xk = qk_w @ s
      AView{F(IN_QK_W), 0, NC, 1},
      LdHalf{s16, (long)NC*NG, NG, nullptr, nullptr, nullptr, 0},
      24, NC, NG, nullptr, xk16, xk32, nullptr, (long)24*NG);
  gemm_k<LdHalf><<<gemm_grid(NG, NG), 128, 0, stream>>>(   // logits = xk^T xk
      AView{xk32, (long)24*NG, 1, NG},
      LdHalf{xk16, (long)24*NG, NG, nullptr, nullptr, nullptr, 0},
      NG, 24, NG, nullptr, attn, nullptr, nullptr, (long)NG*NG);
  softmax_kernel<<<dim3(NG, NB), 256, 0, stream>>>(attn);
  colsum_kernel<<<dim3(NG/256, NB), 256, 0, stream>>>(attn, colsum);
  scaleattn_kernel<<<gs_blocks((long)NB*NG*NG), 256, 0, stream>>>(attn, colsum);
  gemm_k<LdHalf><<<gemm_grid(NC, NG), 128, 0, stream>>>(   // xv = v_w @ s + v_b
      AView{F(IN_V_W), 0, NC, 1},
      LdHalf{s16, (long)NC*NG, NG, nullptr, nullptr, nullptr, 0},
      NC, NC, NG, F(IN_V_B), nullptr, xv32, nullptr, (long)NC*NG);
  gemm_k<LdHalf><<<gemm_grid(NC, NG), 128, 0, stream>>>(   // xr = xv @ attn
      AView{xv32, (long)NC*NG, NG, 1},
      LdHalf{attn, (long)NG*NG, NG, nullptr, nullptr, nullptr, 0},
      NC, NG, NG, nullptr, nullptr, xr32, nullptr, (long)NC*NG);
  diff_kernel<<<gs_blocks((long)NB*NC*NG), 256, 0, stream>>>(s32, xr32, dsr16);
  gemm_k<LdHalf><<<gemm_grid(NC, NG), 128, 0, stream>>>(   // tr_w @ (s - xr) + tr_b
      AView{F(IN_SA_TR_W), 0, NC, 1},
      LdHalf{dsr16, (long)NC*NG, NG, nullptr, nullptr, nullptr, 0},
      NC, NC, NG, F(IN_SA_TR_B), trraw, nullptr, acc + SO_SA, (long)NC*NG);
  FINAL(SO_SA, NC, CNT_BG);

  // 10) out = s + relu(bn(tr))
  final_kernel<<<gs_blocks((long)NB*NC*NG), 256, 0, stream>>>(
      s32, trraw, fin + SO_SA, F(IN_SA_G), F(IN_SA_BE), out);
}
